// LongformerAttention_80058190398169
// MI455X (gfx1250) — matrix operand order repaired and checked by reading, not compiled, not run on hardware
//
#include <hip/hip_runtime.h>
#include <hip/hip_bf16.h>

typedef __attribute__((ext_vector_type(16))) _Float16 v16h;
typedef __attribute__((ext_vector_type(8)))  float    v8f;

#define LTOK 4096   // H*W
#define CIN  256
#define NH   8
#define HD   32
#define N1   768    // 3*C

// Build a v16h A/B fragment from two 16-byte (8-half) contiguous runs.
__device__ __forceinline__ v16h ldfrag(const _Float16* p0, const _Float16* p1) {
  union { int4 i[2]; v16h h; } u;
  u.i[0] = *(const int4*)p0;
  u.i[1] = *(const int4*)p1;
  return u.h;
}

__device__ __forceinline__ bool is_gtok(int i) {
  return i == 0 || i == 63 || i == (LTOK - 64) || i == (LTOK - 1);
}

// ---------------------------------------------------------------------------
// Kernel 1: qkv = x^T @ w1 + b1, scatter into f16 q (scaled), k, v^T buffers.
// Grid: (M/64 = 128, N1/64 = 12), block 128 (4 waves, each a 16x64 strip).
// ---------------------------------------------------------------------------
__global__ __launch_bounds__(128) void qkv_kernel(
    const float* __restrict__ x, const float* __restrict__ w1,
    const float* __restrict__ b1,
    _Float16* __restrict__ qb, _Float16* __restrict__ kbf,
    _Float16* __restrict__ vT) {
  __shared__ __align__(16) _Float16 sA[64][32];   // [token][k]
  __shared__ __align__(16) _Float16 sB[64][32];   // [n][k]  (transposed w1 tile)
  const int tid   = threadIdx.x;
  const int mbase = blockIdx.x * 64;
  const int nbase = blockIdx.y * 64;
  const int wv = tid >> 5, lane = tid & 31, g = lane >> 4, hl = lane & 15;
  const int bblk = mbase >> 12;             // whole 64-row strip is one batch
  const int tokb = mbase & (LTOK - 1);

  // Per-thread staging coordinates: the element index e = (i*128+tid)*4 has
  // i-invariant column (4*tid)&63 and row tid/16 + 8*i -> one base pointer,
  // immediate offsets i*(8*rowstride).
  const int trow = tid >> 4;          // k sub-row 0..7
  const int tcol = (tid * 4) & 63;    // token / n offset, multiple of 4

  v8f acc[4] = {};
  for (int kb = 0; kb < CIN; kb += 32) {
    const float* pA = &x[(bblk * CIN + kb + trow) * LTOK + tokb + tcol];
    const float* pB = &w1[(kb + trow) * N1 + nbase + tcol];
    float4 ta[4], tb[4];
    #pragma unroll
    for (int i = 0; i < 4; ++i) ta[i] = *(const float4*)(pA + i * (8 * LTOK));
    #pragma unroll
    for (int i = 0; i < 4; ++i) tb[i] = *(const float4*)(pB + i * (8 * N1));
    #pragma unroll
    for (int i = 0; i < 4; ++i) {
      int kk = trow + 8 * i;
      sA[tcol + 0][kk] = (_Float16)ta[i].x;
      sA[tcol + 1][kk] = (_Float16)ta[i].y;
      sA[tcol + 2][kk] = (_Float16)ta[i].z;
      sA[tcol + 3][kk] = (_Float16)ta[i].w;
    }
    #pragma unroll
    for (int i = 0; i < 4; ++i) {
      int kk = trow + 8 * i;
      sB[tcol + 0][kk] = (_Float16)tb[i].x;
      sB[tcol + 1][kk] = (_Float16)tb[i].y;
      sB[tcol + 2][kk] = (_Float16)tb[i].z;
      sB[tcol + 3][kk] = (_Float16)tb[i].w;
    }
    __syncthreads();
    v16h af = ldfrag(&sA[wv * 16 + hl][g * 16], &sA[wv * 16 + hl][g * 16 + 8]);
    #pragma unroll
    for (int s = 0; s < 4; ++s) {
      v16h bf = ldfrag(&sB[s * 16 + hl][g * 16], &sB[s * 16 + hl][g * 16 + 8]);
      acc[s] = __builtin_amdgcn_wmma_f32_16x16x32_f16(
          false, af, false, bf, (short)0, acc[s], false, false);
    }
    __syncthreads();
  }

  // Bias + branchless scatter: per-lane base index + per-lane stride, then
  // one add per element (stride HD for q/k row-major, 1 for v transposed).
  const int tokr = tokb + wv * 16 + g * 8;  // rows i are consecutive tokens
  #pragma unroll
  for (int s = 0; s < 4; ++s) {
    int col = nbase + s * 16 + hl;
    float bias = b1[col];
    int which = col % 3, head = (col / 3) & 7, c = col / 24;
    int bhh = bblk * NH + head;
    _Float16* dst = (which == 0) ? qb : (which == 1) ? kbf : vT;
    float scl = (which == 0) ? 0.17677669529663687f : 1.0f;
    int idx_rm = (bhh * LTOK + tokr) * HD + c;   // q,k row-major base
    int idx_tr = (bhh * HD + c) * LTOK + tokr;   // v transposed base
    int idx    = (which == 2) ? idx_tr : idx_rm;
    int stride = (which == 2) ? 1 : HD;
    #pragma unroll
    for (int i = 0; i < 8; ++i) {
      dst[idx] = (_Float16)((acc[s][i] + bias) * scl);
      idx += stride;
    }
  }
}

// ---------------------------------------------------------------------------
// Kernel 2: banded+global attention with online softmax, one wave per
// (bh, 16-query tile). Key tiles processed in pairs (K=32 for the P@V WMMA).
// Grid: (256/4 = 64, B*NH = 16), block 128 (4 independent waves).
// ---------------------------------------------------------------------------
__global__ __launch_bounds__(128) void attn_kernel(
    const _Float16* __restrict__ qb, const _Float16* __restrict__ kbf,
    const _Float16* __restrict__ vT, _Float16* __restrict__ ob) {
  __shared__ __align__(16) _Float16 pl[4][16][32];  // per-wave P tile
  const int tid = threadIdx.x;
  const int wv = tid >> 5, lane = tid & 31, g = lane >> 4, hl = lane & 15;
  const int qt = blockIdx.x * 4 + wv;
  const int bh = blockIdx.y;
  const int qbase = qt * 16;

  // Uniform per-(b,head) bases; everything below indexes with 32-bit offsets.
  const _Float16* qB = qb  + bh * (LTOK * HD);
  const _Float16* kB = kbf + bh * (LTOK * HD);
  const _Float16* vB = vT  + bh * (HD * LTOK);

  // Q fragment (A layout): row = query token, K = head channel.
  const int qoff = (qbase + hl) * HD + g * 16;
  v16h qf = ldfrag(&qB[qoff], &qB[qoff + 8]);

  // Loop-invariant per-lane offset components.
  const int kofs  = hl * HD + g * 16;          // + j0*HD per tile
  const int vofs0 = hl * LTOK + g * 16;        // + j0 per tile
  const int vofs1 = (16 + hl) * LTOK + g * 16;

  v8f o0 = {}, o1 = {};
  float mrow[8], rs[8];
  #pragma unroll
  for (int i = 0; i < 8; ++i) { mrow[i] = -1e30f; rs[i] = 0.f; }

  const bool qg = (qt == 0 || qt == 3 || qt == 252 || qt == 255);

  for (int pt = 0; pt < 128; ++pt) {
    const int j0 = pt * 32;
    bool band = (j0 + 31 >= qbase - 512) && (j0 <= qbase + 527);
    if (!(band || pt <= 1 || pt >= 126 || qg)) continue;

    // Issue ALL global fragment loads up front so loadcnt drains behind the
    // softmax VALU work (the asm dscnt barrier below would otherwise pin the
    // V loads after the LDS round trip).
    const int ka = j0 * HD + kofs;
    v16h kf0 = ldfrag(&kB[ka], &kB[ka + 8]);
    v16h kf1 = ldfrag(&kB[ka + 16 * HD], &kB[ka + 16 * HD + 8]);
    const int va0 = vofs0 + j0, va1 = vofs1 + j0;
    v16h vf0 = ldfrag(&vB[va0], &vB[va0 + 8]);
    v16h vf1 = ldfrag(&vB[va1], &vB[va1 + 8]);

    v8f z = {};
    v8f s0 = __builtin_amdgcn_wmma_f32_16x16x32_f16(false, qf, false, kf0, (short)0, z, false, false);
    v8f s1 = __builtin_amdgcn_wmma_f32_16x16x32_f16(false, qf, false, kf1, (short)0, z, false, false);

    int c0 = j0 + hl, c1 = c0 + 16;
    bool gc0 = is_gtok(c0), gc1 = is_gtok(c1);
    #pragma unroll
    for (int i = 0; i < 8; ++i) {
      int ig = qbase + g * 8 + i;
      bool gi = is_gtok(ig);
      float e0 = s0[i], e1 = s1[i];
      int d0 = ig - c0; if (d0 < 0) d0 = -d0;
      int d1 = ig - c1; if (d1 < 0) d1 = -d1;
      if (!(d0 <= 512 || gi || gc0)) e0 = -1e30f;
      if (!(d1 <= 512 || gi || gc1)) e1 = -1e30f;
      // Row max across the 16 lanes holding this row (xor<16 stays in half-wave).
      float tm = fmaxf(e0, e1);
      tm = fmaxf(tm, __shfl_xor(tm, 1, 32));
      tm = fmaxf(tm, __shfl_xor(tm, 2, 32));
      tm = fmaxf(tm, __shfl_xor(tm, 4, 32));
      tm = fmaxf(tm, __shfl_xor(tm, 8, 32));
      float mn = fmaxf(mrow[i], tm);
      float al = __expf(mrow[i] - mn);
      mrow[i] = mn;
      float p0 = __expf(e0 - mn), p1 = __expf(e1 - mn);
      float ts = p0 + p1;
      ts += __shfl_xor(ts, 1, 32);
      ts += __shfl_xor(ts, 2, 32);
      ts += __shfl_xor(ts, 4, 32);
      ts += __shfl_xor(ts, 8, 32);
      rs[i] = rs[i] * al + ts;
      o0[i] *= al;
      o1[i] *= al;
      // C-layout -> LDS (row-major P tile) for A-fragment reload.
      pl[wv][g * 8 + i][hl]      = (_Float16)p0;
      pl[wv][g * 8 + i][16 + hl] = (_Float16)p1;
    }
    asm volatile("s_wait_dscnt 0" ::: "memory");  // cross-lane LDS RAW
    v16h pf = ldfrag(&pl[wv][hl][g * 16], &pl[wv][hl][g * 16 + 8]);

    o0 = __builtin_amdgcn_wmma_f32_16x16x32_f16(false, pf, false, vf0, (short)0, o0, false, false);
    o1 = __builtin_amdgcn_wmma_f32_16x16x32_f16(false, pf, false, vf1, (short)0, o1, false, false);
  }

  // Normalize and scatter: out channel = c*NH + head (f16 buffer [B][L][C]).
  // Per-row delta is exactly CIN halves -> one pointer + immediate offsets.
  const int b = bh >> 3, head = bh & 7;
  _Float16* prow = ob + (b * LTOK + qbase + g * 8) * CIN + head;
  #pragma unroll
  for (int i = 0; i < 8; ++i) {
    float inv = 1.0f / rs[i];
    prow[i * CIN + hl * NH]        = (_Float16)(o0[i] * inv);
    prow[i * CIN + (16 + hl) * NH] = (_Float16)(o1[i] * inv);
  }
}

// ---------------------------------------------------------------------------
// Kernel 3: out = ob @ w2 + b2, stored NCHW (transpose on store).
// Grid: (M/64 = 128, CIN/64 = 4), block 128.
// ---------------------------------------------------------------------------
__global__ __launch_bounds__(128) void proj_kernel(
    const _Float16* __restrict__ ob, const float* __restrict__ w2,
    const float* __restrict__ b2, float* __restrict__ out) {
  __shared__ __align__(16) _Float16 sB[64][32];   // [n][k]
  const int tid   = threadIdx.x;
  const int mbase = blockIdx.x * 64;
  const int nbase = blockIdx.y * 64;
  const int wv = tid >> 5, lane = tid & 31, g = lane >> 4, hl = lane & 15;
  const int trow = tid >> 4;          // k sub-row 0..7
  const int tcol = (tid * 4) & 63;    // n offset, multiple of 4

  // A rows are contiguous f16 in ob: one per-wave row pointer, kb immediates.
  const _Float16* arow = ob + (mbase + wv * 16 + hl) * CIN;

  v8f acc[4] = {};
  for (int kb = 0; kb < CIN; kb += 32) {
    const float* pB = &w2[(kb + trow) * CIN + nbase + tcol];
    float4 tb[4];
    #pragma unroll
    for (int i = 0; i < 4; ++i) tb[i] = *(const float4*)(pB + i * (8 * CIN));
    #pragma unroll
    for (int i = 0; i < 4; ++i) {
      int kk = trow + 8 * i;
      sB[tcol + 0][kk] = (_Float16)tb[i].x;
      sB[tcol + 1][kk] = (_Float16)tb[i].y;
      sB[tcol + 2][kk] = (_Float16)tb[i].z;
      sB[tcol + 3][kk] = (_Float16)tb[i].w;
    }
    __syncthreads();
    v16h af = ldfrag(&arow[kb + g * 16], &arow[kb + g * 16 + 8]);
    #pragma unroll
    for (int s = 0; s < 4; ++s) {
      v16h bf = ldfrag(&sB[s * 16 + hl][g * 16], &sB[s * 16 + hl][g * 16 + 8]);
      acc[s] = __builtin_amdgcn_wmma_f32_16x16x32_f16(
          false, af, false, bf, (short)0, acc[s], false, false);
    }
    __syncthreads();
  }

  // Store transposed to NCHW: per lane the 8 rows are consecutive tokens.
  int mrow0 = mbase + wv * 16 + g * 8;
  int b = mrow0 >> 12, tok0 = mrow0 & (LTOK - 1);
  #pragma unroll
  for (int s = 0; s < 4; ++s) {
    int col = nbase + s * 16 + hl;
    float bias = b2[col];
    float* dst = out + (b * CIN + col) * LTOK + tok0;
    float4 lo = make_float4(acc[s][0] + bias, acc[s][1] + bias,
                            acc[s][2] + bias, acc[s][3] + bias);
    float4 hi = make_float4(acc[s][4] + bias, acc[s][5] + bias,
                            acc[s][6] + bias, acc[s][7] + bias);
    *(float4*)dst = lo;
    *(float4*)(dst + 4) = hi;
  }
}

extern "C" void kernel_launch(void* const* d_in, const int* in_sizes, int n_in,
                              void* d_out, int out_size, void* d_ws, size_t ws_size,
                              hipStream_t stream) {
  const float* x  = (const float*)d_in[0];
  const float* w1 = (const float*)d_in[1];
  const float* b1 = (const float*)d_in[2];
  const float* w2 = (const float*)d_in[3];
  const float* b2 = (const float*)d_in[4];

  _Float16* ws = (_Float16*)d_ws;
  const size_t seg = (size_t)2 * NH * LTOK * HD;  // 2,097,152 halves = 4 MB
  _Float16* qb = ws;            // [B*NH][L][HD]   q (pre-scaled)
  _Float16* kb = ws + seg;      // [B*NH][L][HD]   k
  _Float16* vT = ws + 2 * seg;  // [B*NH][HD][L]   v transposed
  _Float16* ob = ws + 3 * seg;  // [B][L][C]       attention output (head-interleaved)

  qkv_kernel<<<dim3(128, 12), 128, 0, stream>>>(x, w1, b1, qb, kb, vT);
  attn_kernel<<<dim3(64, 16), 128, 0, stream>>>(qb, kb, vT, ob);
  proj_kernel<<<dim3(128, 4), 128, 0, stream>>>(ob, w2, b2, (float*)d_out);
}
